// MultiLoraLinear_48954037240382
// MI455X (gfx1250) — matrix operand order repaired and checked by reading, not compiled, not run on hardware
//
#include <hip/hip_runtime.h>

// Problem constants (reference: MAX_LORAS=8, OUT=4096, IN=4096, B=4, S=2048)
#define OUT_DIM 4096
#define IN_DIM  4096
#define SEQ     2048

// Tiling
#define BM 128          // C-tile rows (S)
#define BN 128          // C-tile cols (OUT)
#define BK 64           // K-tile (two 16x16x32 WMMA K-steps)
#define LDS_STRIDE 72   // bf16 elems per LDS row: 64+8 pad -> 144B row stride
                        // (36 dwords: 36*r mod 64 tiles all banks, conflict-free b128)

typedef __attribute__((ext_vector_type(16))) __bf16 v16bf;
typedef __attribute__((ext_vector_type(4)))  __bf16 v4bf;
typedef __attribute__((ext_vector_type(8)))  float  v8f;
typedef __attribute__((ext_vector_type(4)))  float  v4f;

union Frag {
    v16bf v;
    uint4 q[2];
};

// Vector f32x4 -> bf16x4 (RNE): lowers to two v_cvt_pk_bf16_f32 on gfx1250.
__device__ __forceinline__ uint2 cvt_pk_bf16x4(v4f f) {
    union { v4bf h; uint2 u; } r;
    r.h = __builtin_convertvector(f, v4bf);
    return r.u;
}

__global__ void __launch_bounds__(256)
multilora_wmma_bf16_kernel(const float* __restrict__ x,
                           const float* __restrict__ weight,
                           const int*   __restrict__ adapter_ids,
                           float*       __restrict__ out) {
    __shared__ unsigned short lA[BM * LDS_STRIDE];   // X tile: [m][k] bf16
    __shared__ unsigned short lB[BN * LDS_STRIDE];   // W tile: [n][k] bf16 (W rows == B cols)

    const int tid   = threadIdx.x;
    const int lane  = tid & 31;
    const int wave  = tid >> 5;      // 0..7
    const int waveM = wave >> 1;     // 0..3 -> 32-row strip
    const int waveN = wave & 1;      // 0..1 -> 64-col strip
    const int lrow  = lane & 15;
    const int lhi   = lane >> 4;     // 0 = lanes 0-15, 1 = lanes 16-31

    const int batch = blockIdx.z;
    const int tileM = blockIdx.y * BM;
    const int tileN = blockIdx.x * BN;

    const int adapter = adapter_ids[batch];
    const float* __restrict__ X = x      + (size_t)batch   * SEQ * IN_DIM;
    const float* __restrict__ W = weight + (size_t)adapter * OUT_DIM * IN_DIM;
    float*       __restrict__ C = out    + (size_t)batch   * SEQ * OUT_DIM;

    // Staging map: thread handles 8 float4 chunks per tile.
    // chunk i -> row (tid>>4) + 16*i, col (tid&15)*4  => single base + immediate offsets
    const int r0 = tid >> 4;
    const int c0 = (tid & 15) << 2;
    const float*    pa0 = X + (size_t)(tileM + r0) * IN_DIM + c0;
    const float*    pb0 = W + (size_t)(tileN + r0) * IN_DIM + c0;
    unsigned short* sA0 = &lA[r0 * LDS_STRIDE + c0];
    unsigned short* sB0 = &lB[r0 * LDS_STRIDE + c0];

    v8f acc[2][4] = {};   // wave tile: 2(M) x 4(N) of 16x16 f32

    // Prologue: fetch first K-tile into registers
    v4f ra[8], rb[8];
#pragma unroll
    for (int i = 0; i < 8; ++i) {
        ra[i] = *(const v4f*)(pa0 + (size_t)i * 16 * IN_DIM);
        rb[i] = *(const v4f*)(pb0 + (size_t)i * 16 * IN_DIM);
    }

    for (int kt = 0; kt < IN_DIM; kt += BK) {
        // ---- Convert previously fetched registers -> LDS (bf16) ----
#pragma unroll
        for (int i = 0; i < 8; ++i) {
            *(uint2*)(sA0 + i * 16 * LDS_STRIDE) = cvt_pk_bf16x4(ra[i]);
            *(uint2*)(sB0 + i * 16 * LDS_STRIDE) = cvt_pk_bf16x4(rb[i]);
        }
        __syncthreads();

        // ---- Issue NEXT tile's global loads; latency hides behind WMMAs ----
        if (kt + BK < IN_DIM) {
#pragma unroll
            for (int i = 0; i < 8; ++i) {
                ra[i] = *(const v4f*)(pa0 + (size_t)i * 16 * IN_DIM + kt + BK);
                rb[i] = *(const v4f*)(pb0 + (size_t)i * 16 * IN_DIM + kt + BK);
            }
        }

        // ---- Two 16x16x32 WMMA K-steps over the staged BK=64 ----
#pragma unroll
        for (int ks = 0; ks < BK; ks += 32) {
            Frag a[2], b[4];

            // A frag (16x32 MxK): lanes 0-15 K=ks..+7 & ks+16..+23; lanes 16-31 +8
#pragma unroll
            for (int mi = 0; mi < 2; ++mi) {
                const int r = waveM * 32 + mi * 16 + lrow;
                const unsigned short* p = &lA[r * LDS_STRIDE + ks + lhi * 8];
                a[mi].q[0] = *(const uint4*)p;          // K ks+0..7   (hi: +8..15)
                a[mi].q[1] = *(const uint4*)(p + 16);   // K ks+16..23 (hi: +24..31)
            }
            // B frag (32x16 KxN): lane n = column N; lanes 0-15 K=ks..+7 & ks+16..+23; lanes 16-31 +8
#pragma unroll
            for (int ni = 0; ni < 4; ++ni) {
                const int cc = waveN * 64 + ni * 16 + lrow;
                const unsigned short* p = &lB[cc * LDS_STRIDE + ks + lhi * 8];
                b[ni].q[0] = *(const uint4*)p;          // K ks+0..7   (hi: +8..15)
                b[ni].q[1] = *(const uint4*)(p + 16);   // K ks+16..23 (hi: +24..31)
            }

#pragma unroll
            for (int mi = 0; mi < 2; ++mi)
#pragma unroll
                for (int ni = 0; ni < 4; ++ni)
                    acc[mi][ni] = __builtin_amdgcn_wmma_f32_16x16x32_bf16(
                        false, a[mi].v, false, b[ni].v,
                        (short)0, acc[mi][ni], false, false);
        }
        __syncthreads();   // all LDS reads done before next overwrite
    }

    // ---- Epilogue: lanes 0-15: M=j, N=lane; lanes 16-31: M=8+j, N=lane-16 ----
#pragma unroll
    for (int mi = 0; mi < 2; ++mi) {
#pragma unroll
        for (int ni = 0; ni < 4; ++ni) {
            const int col = tileN + waveN * 64 + ni * 16 + lrow;
#pragma unroll
            for (int j = 0; j < 8; ++j) {
                const int row = tileM + waveM * 32 + mi * 16 + lhi * 8 + j;
                C[(size_t)row * OUT_DIM + col] = acc[mi][ni][j];
            }
        }
    }
}

extern "C" void kernel_launch(void* const* d_in, const int* in_sizes, int n_in,
                              void* d_out, int out_size, void* d_ws, size_t ws_size,
                              hipStream_t stream) {
    (void)n_in; (void)out_size; (void)d_ws; (void)ws_size;
    const float* x   = (const float*)d_in[0];
    const float* w   = (const float*)d_in[1];
    const int*   ids = (const int*)d_in[2];
    float*       out = (float*)d_out;

    const int B = in_sizes[2];               // adapter_ids has B entries
    dim3 grid(OUT_DIM / BN, SEQ / BM, B);    // 32 x 16 x 4 = 2048 workgroups
    dim3 block(256);                          // 8 wave32 waves
    hipLaunchKernelGGL(multilora_wmma_bf16_kernel, grid, block, 0, stream,
                       x, w, ids, out);
}
